// LSTMCell_77962246357481
// MI455X (gfx1250) — compile-verified, matrix operand order checked by reading, not run on hardware
//
#include <hip/hip_runtime.h>
#include <hip/hip_bf16.h>

// ---------------------------------------------------------------------------
// Coupled-gate LSTM cell for MI455X (gfx1250, wave32, WMMA).
//   B=8192, IN=2048, H=2048, K=IN+H=4096, 3 gates (f, o, c).
//   g = [x|h] @ [Wf;Wo;Wc]^T + b ;  f=sig, o=sig, chat=tanh
//   c_t = f*c_{t-1} + (1-f)*chat ;  h_t = o*tanh(c_t)
// Strategy: fp32 -> bf16 pre-pass into d_ws, then one fused GEMM kernel that
// computes all three gate tiles for the same (b,j) block and applies the
// cell update in registers (no [B,3H] intermediate in memory).
// Wave tile: 64(M) x 32(N) per gate -> 24 v_wmma per 32-wide K step against
// 20 b128 loads; 192 VGPRs of f32 accumulators (wave32 has 1024 VGPRs).
// ---------------------------------------------------------------------------

typedef __attribute__((ext_vector_type(16))) __bf16 v16bf;
typedef __attribute__((ext_vector_type(8)))  __bf16 v8bf;
typedef __attribute__((ext_vector_type(4)))  __bf16 v4bf;
typedef __attribute__((ext_vector_type(8)))  float  v8f;

#define B_DIM  8192
#define IN_DIM 2048
#define H_DIM  2048
#define K_DIM  4096   // IN_DIM + H_DIM
#define NG     3      // gates: f, o, c
#define MI     4      // 16-row tiles per wave along M (64 rows)
#define NI     2      // 16-col tiles per wave along N (32 cols)

// ---- fp32 -> bf16 (round to nearest even) ---------------------------------
__device__ __forceinline__ __bf16 f2bf(float f) {
    unsigned u = __builtin_bit_cast(unsigned, f);
    u += 0x7FFFu + ((u >> 16) & 1u);
    unsigned short s = (unsigned short)(u >> 16);
    return __builtin_bit_cast(__bf16, s);
}

// Build XH = [x_t | h_t_1] in bf16, row-major [B_DIM][K_DIM].
__global__ void build_xh_bf16(const float* __restrict__ x,
                              const float* __restrict__ h,
                              __bf16* __restrict__ xh) {
    size_t i4 = (size_t)blockIdx.x * blockDim.x + threadIdx.x;  // vec4 index
    size_t i  = i4 * 4;
    size_t row = i / K_DIM;
    size_t col = i % K_DIM;                                     // multiple of 4
    const float4* src =
        (col < IN_DIM) ? (const float4*)(x + row * IN_DIM + col)
                       : (const float4*)(h + row * H_DIM + (col - IN_DIM));
    float4 v = *src;
    v4bf o;
    o[0] = f2bf(v.x); o[1] = f2bf(v.y); o[2] = f2bf(v.z); o[3] = f2bf(v.w);
    *(v4bf*)(xh + i) = o;
}

// Plain fp32 -> bf16 convert (for each weight matrix).
__global__ void conv_bf16(const float* __restrict__ src,
                          __bf16* __restrict__ dst) {
    size_t i = ((size_t)blockIdx.x * blockDim.x + threadIdx.x) * 4;
    float4 v = *(const float4*)(src + i);
    v4bf o;
    o[0] = f2bf(v.x); o[1] = f2bf(v.y); o[2] = f2bf(v.z); o[3] = f2bf(v.w);
    *(v4bf*)(dst + i) = o;
}

// ---- fragment load matching the 16-bit WMMA A/B VGPR layout ---------------
// lanes 0-15:  K = {0..7, 16..23}   lanes 16-31: K = {8..15, 24..31}
// (caller pre-offsets the pointer by laneHi*8; chunks are 16B-aligned)
__device__ __forceinline__ v16bf load_frag(const __bf16* __restrict__ p) {
    v8bf lo = *(const v8bf*)(p);
    v8bf hi = *(const v8bf*)(p + 16);
    return __builtin_shufflevector(lo, hi,
        0, 1, 2, 3, 4, 5, 6, 7, 8, 9, 10, 11, 12, 13, 14, 15);
}

__device__ __forceinline__ float fast_sigmoid(float x) {
    return 1.0f / (1.0f + __expf(-x));
}
__device__ __forceinline__ float fast_tanh(float x) {
    // 1 - 2/(e^{2x}+1): saturates cleanly to +/-1 without inf/inf.
    return 1.0f - 2.0f / (__expf(2.0f * x) + 1.0f);
}

// ---- fused GEMM (3 gates) + LSTM cell epilogue ----------------------------
// Block: 256 threads = 8 waves as 2(M) x 4(N). Macro tile 128(M) x 128(N):
// A tile shared by 4 waves, B tiles shared by 2 waves.
__global__ __launch_bounds__(256)
void lstm_gemm_bf16(const __bf16* __restrict__ XH,   // [B_DIM][K_DIM]
                    const __bf16* __restrict__ W,    // [NG][H_DIM][K_DIM]
                    const float*  __restrict__ bF,
                    const float*  __restrict__ bO,
                    const float*  __restrict__ bC,
                    const float*  __restrict__ cPrev, // [B_DIM][H_DIM]
                    float* __restrict__ hOut,         // [B_DIM][H_DIM]
                    float* __restrict__ cOut) {       // [B_DIM][H_DIM]
    const int lane   = threadIdx.x & 31;
    const int wave   = threadIdx.x >> 5;   // 0..7
    const int wm     = wave & 1;           // 2 waves along M
    const int wn     = wave >> 1;          // 4 waves along N
    const int lane16 = lane & 15;
    const int laneHi = lane >> 4;          // 0 or 1

    const int mWave = blockIdx.x * (MI * 16 * 2) + wm * (MI * 16);  // 64-row slab
    const int nWave = blockIdx.y * (NI * 16 * 4) + wn * (NI * 16);  // 32-col slab

    // Lane-resolved base pointers (K-major rows; laneHi*8 = K phase).
    const __bf16* aBase = XH + (size_t)(mWave + lane16) * K_DIM + laneHi * 8;
    const __bf16* bBase = W  + (size_t)(nWave + lane16) * K_DIM + laneHi * 8;

    v8f acc[NG][MI][NI] = {};   // [gate][mi][ni], each a 16x16 f32 tile

    #pragma unroll 1
    for (int k0 = 0; k0 < K_DIM; k0 += 32) {
        // L2-targeted prefetch a few K-steps ahead.
        if (k0 + 256 < K_DIM) {
            __builtin_prefetch(aBase + k0 + 256, 0, 1);
            __builtin_prefetch(bBase + k0 + 256, 0, 1);
        }
        v16bf aF[MI];
        #pragma unroll
        for (int mi = 0; mi < MI; ++mi)
            aF[mi] = load_frag(aBase + (size_t)mi * 16 * K_DIM + k0);

        #pragma unroll
        for (int g = 0; g < NG; ++g) {
            const __bf16* bg = bBase + (size_t)g * H_DIM * K_DIM + k0;
            v16bf bFr[NI];
            #pragma unroll
            for (int ni = 0; ni < NI; ++ni)
                bFr[ni] = load_frag(bg + (size_t)ni * 16 * K_DIM);
            #pragma unroll
            for (int mi = 0; mi < MI; ++mi)
                #pragma unroll
                for (int ni = 0; ni < NI; ++ni)
                    acc[g][mi][ni] = __builtin_amdgcn_wmma_f32_16x16x32_bf16(
                        false, aF[mi], false, bFr[ni],
                        (short)0, acc[g][mi][ni], false, false);
        }
    }

    // Epilogue: C/D layout -> element (VGPR r, lane): M = r + laneHi*8, N = lane16.
    #pragma unroll
    for (int ni = 0; ni < NI; ++ni) {
        const int col  = nWave + ni * 16 + lane16;
        const float bf = bF[col];
        const float bo = bO[col];
        const float bc = bC[col];
        #pragma unroll
        for (int mi = 0; mi < MI; ++mi) {
            const int rowBase = mWave + mi * 16 + laneHi * 8;
            #pragma unroll
            for (int r = 0; r < 8; ++r) {
                const size_t idx = (size_t)(rowBase + r) * H_DIM + col;
                const float f  = fast_sigmoid(acc[0][mi][ni][r] + bf);
                const float o  = fast_sigmoid(acc[1][mi][ni][r] + bo);
                const float ch = fast_tanh   (acc[2][mi][ni][r] + bc);
                const float cp = cPrev[idx];
                const float c  = f * cp + (1.0f - f) * ch;
                const float hh = o * fast_tanh(c);
                hOut[idx] = hh;
                cOut[idx] = c;
            }
        }
    }
}

// ---------------------------------------------------------------------------
extern "C" void kernel_launch(void* const* d_in, const int* in_sizes, int n_in,
                              void* d_out, int out_size, void* d_ws, size_t ws_size,
                              hipStream_t stream) {
    (void)in_sizes; (void)n_in; (void)out_size; (void)ws_size;

    const float* x     = (const float*)d_in[0];   // [B, IN]
    const float* hPrev = (const float*)d_in[1];   // [B, H]
    const float* cPrev = (const float*)d_in[2];   // [B, H]
    const float* Wf    = (const float*)d_in[3];   // [H, K]
    const float* bf    = (const float*)d_in[4];
    const float* Wo    = (const float*)d_in[5];
    const float* bo    = (const float*)d_in[6];
    const float* Wc    = (const float*)d_in[7];
    const float* bc    = (const float*)d_in[8];

    float* hOut = (float*)d_out;                       // h_t first
    float* cOut = hOut + (size_t)B_DIM * H_DIM;        // then c_t

    // Workspace: bf16 XH [B][K] then bf16 W [3][H][K]  (~112 MB total)
    __bf16* XH = reinterpret_cast<__bf16*>(d_ws);
    __bf16* Wb = XH + (size_t)B_DIM * K_DIM;
    const size_t wElems = (size_t)H_DIM * K_DIM;

    // 1) fp32 -> bf16 pre-pass (exact power-of-two sizes, no tails)
    build_xh_bf16<<<((size_t)B_DIM * K_DIM / 4) / 256, 256, 0, stream>>>(x, hPrev, XH);
    conv_bf16   <<<(wElems / 4) / 256, 256, 0, stream>>>(Wf, Wb + 0 * wElems);
    conv_bf16   <<<(wElems / 4) / 256, 256, 0, stream>>>(Wo, Wb + 1 * wElems);
    conv_bf16   <<<(wElems / 4) / 256, 256, 0, stream>>>(Wc, Wb + 2 * wElems);

    // 2) fused 3-gate WMMA GEMM + cell update
    dim3 grid(B_DIM / 128, H_DIM / 128);
    lstm_gemm_bf16<<<grid, 256, 0, stream>>>(XH, Wb, bf, bo, bc, cPrev, hOut, cOut);
}
